// BlockDiagonalChannelAttention_103079215454
// MI455X (gfx1250) — matrix operand order repaired and checked by reading, not compiled, not run on hardware
//
#include <hip/hip_runtime.h>

typedef __attribute__((ext_vector_type(16))) _Float16 v16h;
typedef __attribute__((ext_vector_type(8)))  _Float16 v8h;
typedef __attribute__((ext_vector_type(8)))  float    v8f;

#define NC    128      // channels
#define NP    49       // pixels per patch
#define KPAD  64       // padded K / pixel dimension
#define GP    129      // padded G row stride (floats) -> conflict-free row scans
#define HW    224
#define NPH   7
#define ALPHA_ 0.5f

__global__ __launch_bounds__(256)
void bdca_kernel(const float* __restrict__ x, const float* __restrict__ beta,
                 float* __restrict__ outp, float* __restrict__ scp,
                 float* __restrict__ covp, float* __restrict__ ecp)
{
    __shared__ _Float16 flatA[NC][KPAD];   // 16 KB : flat, row-major, K-padded with 0
    __shared__ _Float16 flatT[KPAD][NC];   // 16 KB : flat^T (B-operand of matmul 2)
    __shared__ float    G[NC * GP];        // 64.5 KB : Gram, padded stride
    __shared__ _Float16 Sp[NC][NC];        // 32 KB : S' = Sc + alpha*cov (f16)
    __shared__ float    rowInvSum[NC], rowMax[NC], mu[NC];

    const int tid  = threadIdx.x;
    const int wave = tid >> 5;
    const int lane = tid & 31;
    const int hl   = lane & 15;
    const int hi   = lane >> 4;            // which half-wave (K split)

    const int patch = blockIdx.x;          // 0..4095 = b*1024 + hh*32 + ww
    const int b  = patch >> 10;
    const int m  = patch & 1023;
    const int hh = m >> 5;
    const int ww = m & 31;

    // x[((b*NC + c)*HW + hh*7 + i)*HW + ww*7 + j]
    const int xbase = ((b * NC) * HW + hh * NPH) * HW + ww * NPH;

    // ---- Phase 0: unfold patch into LDS (f16) + transposed copy, zero K-pad ----
    for (int idx = tid; idx < NC * KPAD; idx += 256) {
        const int c = idx >> 6, k = idx & 63;
        float v = 0.f;
        if (k < NP) {
            const int i = k / NPH, j = k - i * NPH;
            v = x[xbase + c * (HW * HW) + i * HW + j];
        }
        const _Float16 h = (_Float16)v;
        flatA[c][k] = h;
        flatT[k][c] = h;
    }
    __syncthreads();

    // ---- Phase 1: Gram G = flat * flat^T (8x8 tiles of 16x16, K=64 -> 2 WMMA) ----
    // Constant trip count: no per-lane loop-exit mask, lets compiler pipeline.
#pragma unroll 2
    for (int q = 0; q < 8; ++q) {
        const int t = (q << 3) + wave;     // uniform within wave
        const int tm = t >> 3, tn = t & 7;
        const int mrow = tm * 16 + hl;     // A row (lanes 0-15 / 16-31 share M)
        const int nrow = tn * 16 + hl;     // B column n; B^T row == flat row n
        v8f acc = {};
#pragma unroll
        for (int kc = 0; kc < 2; ++kc) {
            const int ks = kc * 32;
            // A 16x32 f16 layout: lanes<16 hold K {0..7,16..23}, lanes>=16 {8..15,24..31}
            const _Float16* ap = &flatA[mrow][ks + (hi ? 8 : 0)];
            v8h a0 = *(const v8h*)(ap);
            v8h a1 = *(const v8h*)(ap + 16);
            v16h av = __builtin_shufflevector(a0, a1, 0,1,2,3,4,5,6,7,8,9,10,11,12,13,14,15);
            // B 32x16 f16 layout: lanes<16 hold K {0..7,16..23}, lanes>=16 {8..15,24..31} of column n
            const _Float16* bp = &flatA[nrow][ks + (hi ? 8 : 0)];
            v8h b0 = *(const v8h*)(bp);
            v8h b1 = *(const v8h*)(bp + 16);
            v16h bv = __builtin_shufflevector(b0, b1, 0,1,2,3,4,5,6,7,8,9,10,11,12,13,14,15);
            acc = __builtin_amdgcn_wmma_f32_16x16x32_f16(false, av, false, bv,
                                                         (short)0, acc, false, false);
        }
        // C/D layout: VGPR v -> row tm*16 + v + 8*hi, col tn*16 + hl
        const int gr = tm * 16 + (hi ? 8 : 0);
        const int gc = tn * 16 + hl;
#pragma unroll
        for (int v = 0; v < 8; ++v) G[(gr + v) * GP + gc] = acc[v];
    }
    __syncthreads();

    // ---- Phase 2: row means and softmax stats (one thread per channel row) ----
    // 4-way split accumulators to break the serial dependence chains.
    if (tid < NC) {
        float s0 = 0.f, s1 = 0.f, s2 = 0.f, s3 = 0.f;
        for (int p = 0; p < 48; p += 4) {
            s0 += (float)flatA[tid][p];
            s1 += (float)flatA[tid][p + 1];
            s2 += (float)flatA[tid][p + 2];
            s3 += (float)flatA[tid][p + 3];
        }
        s0 += (float)flatA[tid][48];
        const float muv = (s0 + s1 + s2 + s3) * (1.f / NP);
        mu[tid] = muv;

        const float* gr = &G[tid * GP];
        float m0 = -3.4e38f, m1 = -3.4e38f, m2 = -3.4e38f, m3 = -3.4e38f;
        for (int d = 0; d < NC; d += 4) {
            m0 = fmaxf(m0, gr[d]);
            m1 = fmaxf(m1, gr[d + 1]);
            m2 = fmaxf(m2, gr[d + 2]);
            m3 = fmaxf(m3, gr[d + 3]);
        }
        const float mx = fmaxf(fmaxf(m0, m1), fmaxf(m2, m3));
        float e0 = 0.f, e1 = 0.f, e2 = 0.f, e3 = 0.f;
        for (int d = 0; d < NC; d += 4) {
            e0 += __expf(gr[d]     - mx);
            e1 += __expf(gr[d + 1] - mx);
            e2 += __expf(gr[d + 2] - mx);
            e3 += __expf(gr[d + 3] - mx);
        }
        rowMax[tid]    = mx;
        rowInvSum[tid] = 1.f / ((e0 + e1) + (e2 + e3));
    }
    __syncthreads();

    // ---- Phase 3: emit Sc & cov (coalesced NT streams), S' = Sc+alpha*cov f16 ----
    {
        const int obase = patch << 14;     // patch * 128 * 128
        for (int idx = tid; idx < NC * NC; idx += 256) {
            const int c = idx >> 7, d = idx & 127;
            const float g  = G[c * GP + d];
            const float sc = __expf(g - rowMax[c]) * rowInvSum[c];
            const float cv = g * (1.f / NP) - mu[c] * mu[d];   // cov = G/P - mu mu^T
            __builtin_nontemporal_store(sc, scp  + obase + idx);
            __builtin_nontemporal_store(cv, covp + obase + idx);
            Sp[c][d] = (_Float16)(sc + ALPHA_ * cv);
        }
    }
    __syncthreads();

    // ---- Phase 4: Ec = S' * flat (8x4 tiles, K=128 -> 4 WMMA); fold + outputs ----
    const float betaV = beta[0];
#pragma unroll 2
    for (int q = 0; q < 4; ++q) {
        const int t = (q << 3) + wave;     // {wave, wave+8, wave+16, wave+24} < 32
        const int tm = t >> 2, tn = t & 3;
        const int mrow = tm * 16 + hl;
        const int ncol = tn * 16 + hl;     // pixel index (0..63)
        v8f acc = {};
#pragma unroll
        for (int kc = 0; kc < 4; ++kc) {
            const int ks = kc * 32;
            const _Float16* ap = &Sp[mrow][ks + (hi ? 8 : 0)];
            v8h a0 = *(const v8h*)(ap);
            v8h a1 = *(const v8h*)(ap + 16);
            v16h av = __builtin_shufflevector(a0, a1, 0,1,2,3,4,5,6,7,8,9,10,11,12,13,14,15);
            const _Float16* bp = &flatT[ncol][ks + (hi ? 8 : 0)];
            v8h b0 = *(const v8h*)(bp);
            v8h b1 = *(const v8h*)(bp + 16);
            v16h bv = __builtin_shufflevector(b0, b1, 0,1,2,3,4,5,6,7,8,9,10,11,12,13,14,15);
            acc = __builtin_amdgcn_wmma_f32_16x16x32_f16(false, av, false, bv,
                                                         (short)0, acc, false, false);
        }
        const int p = tn * 16 + hl;        // divergence only after the WMMAs
        if (p < NP) {
            const int i = p / NPH, j = p - i * NPH;
            const int cbase = tm * 16 + (hi ? 8 : 0);
#pragma unroll
            for (int v = 0; v < 8; ++v) {
                const int c = cbase + v;
                const int g = xbase + c * (HW * HW) + i * HW + j;
                const float xv = x[g];
                const float ec = acc[v];
                __builtin_nontemporal_store(ec, ecp + g);                    // Ec_map
                __builtin_nontemporal_store(xv * (betaV * ec + xv), outp + g); // out
            }
        }
    }
}

extern "C" void kernel_launch(void* const* d_in, const int* in_sizes, int n_in,
                              void* d_out, int out_size, void* d_ws, size_t ws_size,
                              hipStream_t stream) {
    (void)in_sizes; (void)n_in; (void)d_ws; (void)ws_size; (void)out_size;
    const float* x    = (const float*)d_in[0];
    const float* beta = (const float*)d_in[1];
    float* out = (float*)d_out;
    const long long N1 = 4LL * 128 * 224 * 224;   // 25,690,112 (out / Ec_map)
    const long long N2 = 4LL * 1024 * 128 * 128;  // 67,108,864 (Sc / cov)
    float* sc  = out + N1;
    float* cov = out + N1 + N2;
    float* ec  = out + N1 + 2 * N2;
    bdca_kernel<<<4096, 256, 0, stream>>>(x, beta, out, sc, cov, ec);
}
